// EdgeMLP_20486994002503
// MI455X (gfx1250) — matrix operand order repaired and checked by reading, not compiled, not run on hardware
//
#include <hip/hip_runtime.h>
#include <hip/hip_bf16.h>

typedef __attribute__((ext_vector_type(16))) __bf16 v16bf;
typedef __attribute__((ext_vector_type(8)))  __bf16 v8bf;
typedef __attribute__((ext_vector_type(4)))  __bf16 v4bf;
typedef __attribute__((ext_vector_type(8)))  float  v8f;
typedef __attribute__((ext_vector_type(4)))  float  v4f;

#define SA 520                 // LDS activation row stride (bf16 elems): 512 + 8 pad (banks)
#define LDS_ACT (128 * SA)     // one activation buffer, halves
#define PANEL_HALVES 4096      // one 64(K) x 64(N) bf16 weight panel

union AV { v16bf v; v8bf h[2]; };

#define WMMA_BF16(A, B, C) \
    __builtin_amdgcn_wmma_f32_16x16x32_bf16(false, (A), false, (B), (short)0, (C), false, false)

// SiLU with hardware reciprocal (v_rcp_f32, ~1 ulp): far more accurate than the
// bf16 quantization already present in the matmul inputs, and 1 TRANS op
// instead of a full IEEE divide expansion.
__device__ __forceinline__ float silu_fast(float x)
{
#if __has_builtin(__builtin_amdgcn_rcpf)
    return x * __builtin_amdgcn_rcpf(1.0f + __expf(-x));
#else
    return x / (1.0f + __expf(-x));
#endif
}

// ---------------------------------------------------------------------------
// Weight pre-swizzle: f32 row-major W[K][Nout] -> bf16 fragment-ready layout.
// Panel p = nc*(K/64)+kp is 4096 contiguous halves. Within a panel:
//   idx = ((kk*4 + ns)*32 + lane)*16 + e
//   k = kp*64 + kk*32 + (lane>=16 ? 16 : 0) + e     (B frag: lanes split K at 16)
//   n = nc*64 + ns*16 + (lane & 15)
// ---------------------------------------------------------------------------
__global__ void __launch_bounds__(256, 1)
edge_mlp_wswz(const float* __restrict__ W, __bf16* __restrict__ dst, int K, int Nout)
{
    int idx = blockIdx.x * 256 + threadIdx.x;
    if (idx >= K * Nout) return;
    int kpanels = K >> 6;
    int p    = idx >> 12;
    int w    = idx & 4095;
    int kp   = p % kpanels;
    int nc   = p / kpanels;
    int kk   = w >> 11;
    int rem  = w & 2047;
    int ns   = rem >> 9;
    int rem2 = rem & 511;
    int ln   = rem2 >> 4;
    int e    = rem2 & 15;
    int k = (kp << 6) + (kk << 5) + ((ln >> 4) << 4) + e;
    int n = (nc << 6) + (ns << 4) + (ln & 15);
    dst[idx] = (__bf16)W[k * Nout + n];
}

// ---------------------------------------------------------------------------
// Async copy of one 8 KB weight panel: global -> LDS without VGPR round-trip.
// 256 threads x 32 B each, two GLOBAL_LOAD_ASYNC_TO_LDS_B128 per thread
// (INST_OFFSET is added on both the LDS and global sides per ISA 10.7/15.18).
// Tracked by ASYNCcnt; publisher does s_wait_asynccnt 0 before the barrier.
// ---------------------------------------------------------------------------
__device__ __forceinline__ void stage_panel_async(__bf16* ldsDst,
                                                  const __bf16* gSrc, int tid)
{
    unsigned lds = (unsigned)(size_t)(ldsDst) + (unsigned)tid * 32u; // LDS byte addr
    const char* g = (const char*)gSrc + (size_t)tid * 32u;
    asm volatile("global_load_async_to_lds_b128 %0, %1, off\n\t"
                 "global_load_async_to_lds_b128 %0, %1, off offset:16"
                 :: "v"(lds), "v"(g) : "memory");
}

__device__ __forceinline__ void wait_async_all()
{
    asm volatile("s_wait_asynccnt 0x0" ::: "memory");
}

// ---------------------------------------------------------------------------
// Accumulate one 64-col N-chunk over full K for this wave's 16-row M-subtile.
// Weight panels (64x64) async-staged in LDS, double-buffered.
// ---------------------------------------------------------------------------
template <int KDIM>
__device__ __forceinline__ void gemm_nchunk(const __bf16* __restrict__ actIn,
                                            __bf16* __restrict__ Wbuf,
                                            const __bf16* __restrict__ wlayer,
                                            int nc, int msub, int lane, int tid,
                                            v8f acc[4])
{
    constexpr int KP = KDIM / 64;
    // stage panel 0
    stage_panel_async(Wbuf, wlayer + (size_t)(nc * KP) * PANEL_HALVES, tid);
    wait_async_all();
    __syncthreads();

    const int arow = (msub * 16 + (lane & 15)) * SA + ((lane >> 4) << 4);

#pragma unroll 1
    for (int kp = 0; kp < KP; ++kp) {
        const bool more = (kp + 1 < KP);
        if (more)  // kick next panel's async copy; overlaps with the WMMAs below
            stage_panel_async(Wbuf + ((kp + 1) & 1) * PANEL_HALVES,
                              wlayer + (size_t)(nc * KP + kp + 1) * PANEL_HALVES, tid);

        const __bf16* P = Wbuf + (kp & 1) * PANEL_HALVES;

        // Batch ALL fragment loads (2 A + 8 B = 20 ds_load_b128) before the
        // 8-WMMA chain: partial dscnt waits, matrix pipe runs uninterrupted.
        const __bf16* ap = actIn + arow + kp * 64;
        AV a0, a1;
        a0.h[0] = *(const v8bf*)(ap);          // kk=0: e0..7  -> K kbase+0..7
        a0.h[1] = *(const v8bf*)(ap + 8);      // kk=0: e8..15 -> K kbase+8..15
        a1.h[0] = *(const v8bf*)(ap + 32);     // kk=1
        a1.h[1] = *(const v8bf*)(ap + 40);
        const __bf16* bp = P + lane * 16;
        v16bf b0 = *(const v16bf*)(bp);
        v16bf b1 = *(const v16bf*)(bp + 512);
        v16bf b2 = *(const v16bf*)(bp + 1024);
        v16bf b3 = *(const v16bf*)(bp + 1536);
        v16bf b4 = *(const v16bf*)(bp + 2048);
        v16bf b5 = *(const v16bf*)(bp + 2560);
        v16bf b6 = *(const v16bf*)(bp + 3072);
        v16bf b7 = *(const v16bf*)(bp + 3584);

        acc[0] = WMMA_BF16(a0.v, b0, acc[0]);
        acc[1] = WMMA_BF16(a0.v, b1, acc[1]);
        acc[2] = WMMA_BF16(a0.v, b2, acc[2]);
        acc[3] = WMMA_BF16(a0.v, b3, acc[3]);
        acc[0] = WMMA_BF16(a1.v, b4, acc[0]);
        acc[1] = WMMA_BF16(a1.v, b5, acc[1]);
        acc[2] = WMMA_BF16(a1.v, b6, acc[2]);
        acc[3] = WMMA_BF16(a1.v, b7, acc[3]);

#if __has_builtin(__builtin_amdgcn_sched_group_barrier)
        // Force the schedule: 20 DS reads first, then the 8 WMMAs.
        // 0x100 = DS read, 0x8 = MFMA/WMMA.
        __builtin_amdgcn_sched_group_barrier(0x100, 20, 0);
        __builtin_amdgcn_sched_group_barrier(0x008, 8, 0);
#endif

        if (more) wait_async_all();
        __syncthreads();
    }
}

template <int KDIM>
__device__ __forceinline__ void gemm_silu_layer(const __bf16* __restrict__ actIn,
                                                __bf16* __restrict__ actOut,
                                                __bf16* __restrict__ Wbuf,
                                                const __bf16* __restrict__ wlayer,
                                                const float* __restrict__ bias,
                                                int msub, int lane, int tid)
{
#pragma unroll 1
    for (int nc = 0; nc < 8; ++nc) {   // Nout = 512 in chunks of 64
        v8f acc[4];
#pragma unroll
        for (int ns = 0; ns < 4; ++ns) {
            float bv = bias[nc * 64 + ns * 16 + (lane & 15)];
#pragma unroll
            for (int i = 0; i < 8; ++i) acc[ns][i] = bv;
        }
        gemm_nchunk<KDIM>(actIn, Wbuf, wlayer, nc, msub, lane, tid, acc);
        // SiLU + bf16 store to LDS. C/D layout: elem r -> row r + (lane>=16 ? 8:0)
#pragma unroll
        for (int ns = 0; ns < 4; ++ns) {
            int n = nc * 64 + ns * 16 + (lane & 15);
#pragma unroll
            for (int r = 0; r < 8; ++r) {
                float s = silu_fast(acc[ns][r]);
                actOut[(msub * 16 + r + ((lane >> 4) << 3)) * SA + n] = (__bf16)s;
            }
        }
    }
}

// ---------------------------------------------------------------------------
// Fused edge MLP: gather+concat -> L1+SiLU -> L2+SiLU -> L3 -> LayerNorm
// One workgroup (8 waves) handles 128 edges; wave w owns rows 16w..16w+15.
// ---------------------------------------------------------------------------
__global__ void __launch_bounds__(256, 1)
edge_mlp_fused(const float* __restrict__ efeat, const float* __restrict__ nfeat,
               const int* __restrict__ srcI, const int* __restrict__ dstI,
               const float* __restrict__ b1, const float* __restrict__ b2,
               const float* __restrict__ b3, const float* __restrict__ gamma,
               const float* __restrict__ beta, const __bf16* __restrict__ wsw,
               float* __restrict__ out, int E)
{
    extern __shared__ __align__(32) __bf16 sm[];
    __bf16* Abuf = sm;                  // 128 x SA: cat, then h2
    __bf16* Hbuf = sm + LDS_ACT;        // 128 x SA: h1
    __bf16* Wbuf = sm + 2 * LDS_ACT;    // 2 x 4096: weight panel ping-pong

    const int tid   = threadIdx.x;
    const int lane  = tid & 31;
    const int msub  = tid >> 5;
    const int edge0 = blockIdx.x * 128;

    // ---- gather cat = [efeat | nfeat[src] | nfeat[dst]] for this wave's 16 rows
#pragma unroll 1
    for (int r = 0; r < 16; ++r) {
        int e = edge0 + msub * 16 + r;
        int rowoff = (msub * 16 + r) * SA;
        if (e < E) {
            const float* bases[3] = { efeat + (size_t)e * 128,
                                      nfeat + (size_t)srcI[e] * 128,
                                      nfeat + (size_t)dstI[e] * 128 };
#pragma unroll
            for (int j = 0; j < 3; ++j) {
                v4f x = *(const v4f*)(bases[j] + lane * 4);
                v4bf y;
                y[0] = (__bf16)x[0]; y[1] = (__bf16)x[1];
                y[2] = (__bf16)x[2]; y[3] = (__bf16)x[3];
                *(v4bf*)(Abuf + rowoff + j * 128 + lane * 4) = y;
            }
        } else {
            v4bf z = { (__bf16)0.f, (__bf16)0.f, (__bf16)0.f, (__bf16)0.f };
#pragma unroll
            for (int j = 0; j < 3; ++j)
                *(v4bf*)(Abuf + rowoff + j * 128 + lane * 4) = z;
        }
    }
    // A rows are wave-private (same wave writes & reads); Wbuf barriers follow.

    // ---- layer 1: [128,384] x W1[384,512] + b1, SiLU
    gemm_silu_layer<384>(Abuf, Hbuf, Wbuf, wsw, b1, msub, lane, tid);
    // ---- layer 2: [128,512] x W2[512,512] + b2, SiLU  (h2 overwrites cat buf)
    gemm_silu_layer<512>(Hbuf, Abuf, Wbuf, wsw + 384 * 512, b2, msub, lane, tid);

    // ---- layer 3: [128,512] x W3[512,128] + b3, kept in registers (16x128/wave)
    v8f acc3[2][4];
    const __bf16* w3 = wsw + 384 * 512 + 512 * 512;
#pragma unroll 1
    for (int nc = 0; nc < 2; ++nc) {
#pragma unroll
        for (int ns = 0; ns < 4; ++ns) {
            float bv = b3[nc * 64 + ns * 16 + (lane & 15)];
#pragma unroll
            for (int i = 0; i < 8; ++i) acc3[nc][ns][i] = bv;
        }
        gemm_nchunk<512>(Abuf, Wbuf, w3, nc, msub, lane, tid, acc3[nc]);
    }

    // ---- LayerNorm over 128 cols per row. Row r's cols live in one 16-lane half
    // (lanes 0-15 hold row r, lanes 16-31 hold row r+8) -> shfl_xor within 16.
#pragma unroll
    for (int r = 0; r < 8; ++r) {
        float s = 0.f, q = 0.f;
#pragma unroll
        for (int nc = 0; nc < 2; ++nc)
#pragma unroll
            for (int ns = 0; ns < 4; ++ns) {
                float x = acc3[nc][ns][r];
                s += x; q += x * x;
            }
#pragma unroll
        for (int m = 1; m < 16; m <<= 1) {
            s += __shfl_xor(s, m, 32);
            q += __shfl_xor(q, m, 32);
        }
        float mu  = s * (1.0f / 128.0f);
        float var = q * (1.0f / 128.0f) - mu * mu;
        float inv = rsqrtf(var + 1e-5f);

        int row = edge0 + msub * 16 + r + ((lane >> 4) << 3);
        if (row < E) {
            float* op = out + (size_t)row * 128;
#pragma unroll
            for (int nc = 0; nc < 2; ++nc)
#pragma unroll
                for (int ns = 0; ns < 4; ++ns) {
                    int n = nc * 64 + ns * 16 + (lane & 15);
                    op[n] = (acc3[nc][ns][r] - mu) * inv * gamma[n] + beta[n];
                }
        }
    }
}

// ---------------------------------------------------------------------------
extern "C" void kernel_launch(void* const* d_in, const int* in_sizes, int n_in,
                              void* d_out, int out_size, void* d_ws, size_t ws_size,
                              hipStream_t stream)
{
    const float* efeat = (const float*)d_in[0];
    const float* nfeat = (const float*)d_in[1];
    const int*   srcI  = (const int*)d_in[2];
    const int*   dstI  = (const int*)d_in[3];
    const float* W1    = (const float*)d_in[4];
    const float* b1    = (const float*)d_in[5];
    const float* W2    = (const float*)d_in[6];
    const float* b2    = (const float*)d_in[7];
    const float* W3    = (const float*)d_in[8];
    const float* b3    = (const float*)d_in[9];
    const float* gamma = (const float*)d_in[10];
    const float* beta  = (const float*)d_in[11];
    float* out = (float*)d_out;

    const int E = in_sizes[0] / 128;   // efeat is [E,128]

    // bf16 swizzled weights in workspace: 384*512 + 512*512 + 512*128 halves = 1 MB
    __bf16* ws1 = (__bf16*)d_ws;
    __bf16* ws2 = ws1 + 384 * 512;
    __bf16* ws3 = ws2 + 512 * 512;

    edge_mlp_wswz<<<(384 * 512) / 256, 256, 0, stream>>>(W1, ws1, 384, 512);
    edge_mlp_wswz<<<(512 * 512) / 256, 256, 0, stream>>>(W2, ws2, 512, 512);
    edge_mlp_wswz<<<(512 * 128) / 256, 256, 0, stream>>>(W3, ws3, 512, 128);

    int blocks = (E + 127) / 128;
    size_t smem = (size_t)(2 * LDS_ACT + 2 * PANEL_HALVES) * sizeof(__bf16); // 282,624 B
    edge_mlp_fused<<<blocks, 256, smem, stream>>>(efeat, nfeat, srcI, dstI,
                                                  b1, b2, b3, gamma, beta,
                                                  ws1, out, E);
}
